// LAS_66142496358834
// MI455X (gfx1250) — compile-verified, matrix operand order checked by reading, not run on hardware
//
#include <hip/hip_runtime.h>
#include <math.h>

// ---------------- types ----------------
typedef __bf16 bf16_t;
typedef bf16_t v16bf __attribute__((ext_vector_type(16)));
typedef float  v8f   __attribute__((ext_vector_type(8)));

#define NWG   64
#define TPB   256
#define GTH   (NWG * TPB)
#define NWAVE (NWG * (TPB / 32))

#define Hs   512
#define Bs   64
#define Ts   1024
#define Ls   256
#define Vs   34
#define SCALEF 0.08838834764831845f   // 1/sqrt(128)

// A-panel index: value = h[row = mt*16 + (lane&15), k = kt*32 + ((lane>>4)<<4) + j]
__device__ __forceinline__ size_t aidx(int b, int k, int Kt) {
  int mt = b >> 4;
  int kt = k >> 5;
  int lane = (b & 15) | (((k >> 4) & 1) << 4);
  int j = k & 15;
  return ((size_t)(mt * Kt + kt) * 32 + lane) * 16 + j;
}

__device__ __forceinline__ float sigf(float x) { return 1.0f / (1.0f + __expf(-x)); }

// ---------------- grid-wide spin barrier ----------------
__device__ __forceinline__ void grid_barrier(unsigned* cnt, unsigned* sRound) {
  __syncthreads();
  if (threadIdx.x == 0) {
    unsigned r = ++(*sRound);
    __threadfence();
    atomicAdd(cnt, 1u);
    unsigned target = r * (unsigned)NWG;
    while (atomicAdd(cnt, 0u) < target) { __builtin_amdgcn_s_sleep(2); }
    __threadfence();
  }
  __syncthreads();
}

// ---------------- WMMA bf16 GEMM:  D[64,N] = A[64,K] * B^T  (+Cin) ----------------
// A packed panels (per (mt,kt): 32 lanes x 16 bf16 contiguous), B packed likewise.
__device__ __forceinline__ void gemm_bf16(
    const bf16_t* __restrict__ A, const bf16_t* __restrict__ B,
    const float* __restrict__ Cin, float* __restrict__ D,
    int N, int Kt, int gw, int lane, bool lrelu)
{
  int ntiles = N >> 4;
  int total = ntiles * 4;            // 4 M-tiles (M=64)
  for (int tile = gw; tile < total; tile += NWAVE) {
    int mt = tile & 3;
    int nt = tile >> 2;
    int row0 = mt * 16 + (lane >> 4) * 8;
    int col  = nt * 16 + (lane & 15);
    v8f acc;
    if (Cin) {
      #pragma unroll
      for (int r = 0; r < 8; ++r) acc[r] = Cin[(size_t)(row0 + r) * N + col];
    } else {
      #pragma unroll
      for (int r = 0; r < 8; ++r) acc[r] = 0.0f;
    }
    const bf16_t* Ap = A + (size_t)mt * Kt * 512 + lane * 16;
    const bf16_t* Bp = B + (size_t)nt * Kt * 512 + lane * 16;
    for (int kt = 0; kt < Kt; ++kt) {
      if (kt + 2 < Kt) __builtin_prefetch(Bp + (size_t)(kt + 2) * 512, 0, 1);
      v16bf a = *(const v16bf*)(Ap + (size_t)kt * 512);
      v16bf b = *(const v16bf*)(Bp + (size_t)kt * 512);
      acc = __builtin_amdgcn_wmma_f32_16x16x32_bf16(false, a, false, b, (short)0,
                                                    acc, false, false);
    }
    #pragma unroll
    for (int r = 0; r < 8; ++r) {
      float v = acc[r];
      if (lrelu) v = (v > 0.0f) ? v : 0.01f * v;
      D[(size_t)(row0 + r) * N + col] = v;
    }
  }
}

// ---------------- persistent kernel args ----------------
struct PArgs {
  const bf16_t *Whh0p, *Whh1p, *Wih1p, *Wih2p, *Whh2p, *Wp1aP;
  float *G0, *G1, *G2;
  bf16_t *hA0, *hA1, *hA2;
  float *c0, *c1, *c2;
  float *yF;
  const float *embT, *ctxPart, *ctxProj;
  const float *bih1, *bhh1, *bih2, *bhh2;
  const float *Wemb, *bproj2;
  const int *label;
  float *out;
  unsigned *cnt;
};

// ---------------- cell phases (elementwise) ----------------
__device__ void phase_cell0(const PArgs& a, int step, int gtid) {
  for (int e = gtid; e < 64 * 1024; e += GTH) {
    int b = e >> 10, j = e & 1023;
    int lab = a.label[b * Ls + step];
    const float* g  = a.G0      + (size_t)b   * 4096;
    const float* et = a.embT    + (size_t)lab * 4096;
    const float* cp = a.ctxPart + (size_t)b   * 4096;
    float gi = g[j]        + et[j]        + cp[j];
    float gf = g[j + 1024] + et[j + 1024] + cp[j + 1024];
    float gg = g[j + 2048] + et[j + 2048] + cp[j + 2048];
    float go = g[j + 3072] + et[j + 3072] + cp[j + 3072];
    float c = sigf(gf) * a.c0[e] + sigf(gi) * tanhf(gg);
    a.c0[e] = c;
    a.hA0[aidx(b, j, 32)] = (bf16_t)(sigf(go) * tanhf(c));
  }
}

__device__ void phase_cell1(const PArgs& a, int gtid) {
  for (int e = gtid; e < 64 * 1024; e += GTH) {
    int b = e >> 10, j = e & 1023;
    const float* g = a.G1 + (size_t)b * 4096;
    float gi = g[j]        + a.bih1[j]        + a.bhh1[j];
    float gf = g[j + 1024] + a.bih1[j + 1024] + a.bhh1[j + 1024];
    float gg = g[j + 2048] + a.bih1[j + 2048] + a.bhh1[j + 2048];
    float go = g[j + 3072] + a.bih1[j + 3072] + a.bhh1[j + 3072];
    float c = sigf(gf) * a.c1[e] + sigf(gi) * tanhf(gg);
    a.c1[e] = c;
    a.hA1[aidx(b, j, 32)] = (bf16_t)(sigf(go) * tanhf(c));
  }
}

__device__ void phase_cell2(const PArgs& a, int gtid) {
  for (int e = gtid; e < 64 * 512; e += GTH) {
    int b = e >> 9, j = e & 511;
    const float* g = a.G2 + (size_t)b * 2048;
    float gi = g[j]        + a.bih2[j]        + a.bhh2[j];
    float gf = g[j + 512]  + a.bih2[j + 512]  + a.bhh2[j + 512];
    float gg = g[j + 1024] + a.bih2[j + 1024] + a.bhh2[j + 1024];
    float go = g[j + 1536] + a.bih2[j + 1536] + a.bhh2[j + 1536];
    float c = sigf(gf) * a.c2[e] + sigf(gi) * tanhf(gg);
    a.c2[e] = c;
    a.hA2[aidx(b, j, 16)] = (bf16_t)(sigf(go) * tanhf(c));
  }
}

__device__ void phase_logits(const PArgs& a, int t, float* sBuf) {
  int b = blockIdx.x;      // NWG == B == 64
  const float* y = a.yF + (size_t)b * 512;
  int v = threadIdx.x;
  if (v < Vs) {
    float s = a.bproj2[v];
    const float* w = a.Wemb + (size_t)v * 512;
    for (int h = 0; h < 512; ++h) s += y[h] * w[h];
    sBuf[v] = s;
  }
  __syncthreads();
  if (threadIdx.x == 0) {
    float mx = -1e30f;
    for (int k = 0; k < Vs; ++k) mx = fmaxf(mx, sBuf[k]);
    float sum = 0.0f;
    for (int k = 0; k < Vs; ++k) sum += __expf(sBuf[k] - mx);
    sBuf[Vs] = mx + __logf(sum);
  }
  __syncthreads();
  if (v < Vs) a.out[((size_t)b * 255 + t) * Vs + v] = sBuf[v] - sBuf[Vs];
  __syncthreads();
}

// ---------------- persistent recurrent kernel ----------------
__global__ __launch_bounds__(TPB) void las_persistent(PArgs a) {
  __shared__ unsigned sRound;
  __shared__ float sBuf[Vs + 2];
  if (threadIdx.x == 0) sRound = 0;
  __syncthreads();
  int lane = threadIdx.x & 31;
  int gw   = blockIdx.x * (TPB / 32) + (threadIdx.x >> 5);
  int gtid = blockIdx.x * TPB + threadIdx.x;

  for (int i = 0; i < 256; ++i) {
    // Phase G: hidden projections for step i (uses h from step i-1),
    // plus the output projection y for step i-1 (uses h2 from step i-1).
    if (i < 255) {
      gemm_bf16(a.hA0, a.Whh0p, nullptr, a.G0, 4096, 32, gw, lane, false);
      gemm_bf16(a.hA1, a.Whh1p, nullptr, a.G1, 4096, 32, gw, lane, false);
      gemm_bf16(a.hA2, a.Whh2p, nullptr, a.G2, 2048, 16, gw, lane, false);
    }
    if (i > 0) {
      gemm_bf16(a.hA2, a.Wp1aP, a.ctxProj, a.yF, 512, 16, gw, lane, true);
    }
    grid_barrier(a.cnt, &sRound);

    if (i < 255) phase_cell0(a, i, gtid);
    if (i > 0)   phase_logits(a, i - 1, sBuf);
    if (i == 255) break;
    grid_barrier(a.cnt, &sRound);

    gemm_bf16(a.hA0, a.Wih1p, a.G1, a.G1, 4096, 32, gw, lane, false);
    grid_barrier(a.cnt, &sRound);
    phase_cell1(a, gtid);
    grid_barrier(a.cnt, &sRound);

    gemm_bf16(a.hA1, a.Wih2p, a.G2, a.G2, 2048, 32, gw, lane, false);
    grid_barrier(a.cnt, &sRound);
    phase_cell2(a, gtid);
    grid_barrier(a.cnt, &sRound);
  }
}

// ---------------- precompute kernels ----------------
__global__ void k_init(unsigned* cnt) { if (threadIdx.x == 0) *cnt = 0u; }

// pack fp32 weight [N, stride] (cols colOff..colOff+K) into bf16 WMMA B-panels
__global__ void k_pack(const float* __restrict__ src, bf16_t* __restrict__ dst,
                       int N, int K, int stride, int colOff) {
  int Kt = K >> 5;
  size_t total = (size_t)N * K;
  for (size_t e = (size_t)blockIdx.x * blockDim.x + threadIdx.x; e < total;
       e += (size_t)gridDim.x * blockDim.x) {
    int j = (int)(e & 15);
    int lane = (int)((e >> 4) & 31);
    size_t pair = e >> 9;
    int kt = (int)(pair % Kt);
    int nt = (int)(pair / Kt);
    int n = nt * 16 + (lane & 15);
    int k = kt * 32 + ((lane >> 4) << 4) + j;
    dst[e] = (bf16_t)src[(size_t)n * stride + colOff + k];
  }
}

// query -> nq -> folded key weights wkq / ck   (one block, 512 threads)
__global__ void k_attn_small(const float* h02, const float* Wquery, const float* bquery,
                             const float* Wq, const float* bq,
                             const float* Wk, const float* bk,
                             float* wkq, float* ck) {
  __shared__ float sQ[512];
  __shared__ float sNq[512];
  int tid = threadIdx.x;
  {
    float s = bquery[tid];
    const float* w = Wquery + (size_t)tid * 512;
    for (int h = 0; h < 512; ++h) s += h02[h] * w[h];
    sQ[tid] = s;
  }
  __syncthreads();
  {
    float s = bq[tid];
    const float* w = Wq + (size_t)tid * 512;   // Wq[n][d][h], tid = n*128+d
    for (int h = 0; h < 512; ++h) s += sQ[h] * w[h];
    sNq[tid] = s;
  }
  __syncthreads();
  for (int idx = tid; idx < 2048; idx += 512) {
    int n = idx >> 9, h = idx & 511;
    float s = 0.0f;
    const float* w = Wk + (size_t)n * 128 * 512 + h;
    for (int d = 0; d < 128; ++d) s += sNq[n * 128 + d] * w[(size_t)d * 512];
    wkq[idx] = s;
  }
  if (tid < 4) {
    float s = 0.0f;
    for (int d = 0; d < 128; ++d) s += sNq[tid * 128 + d] * bk[tid * 128 + d];
    ck[tid] = s;
  }
}

// scores[b][n][t] = SCALE * (keys[t,b,:] . wkq[n,:] + ck[n]); one wave per (t,b)
__global__ void k_scores(const float* __restrict__ keys, const float* __restrict__ wkq,
                         const float* __restrict__ ck, float* __restrict__ scores) {
  int wid = blockIdx.x * (blockDim.x / 32) + (threadIdx.x >> 5);
  int lane = threadIdx.x & 31;
  int t = wid >> 6;
  int b = wid & 63;
  const float* kr = keys + ((size_t)t * 64 + b) * 512 + lane * 16;
  const float* w0 = wkq + lane * 16;
  float p0 = 0, p1 = 0, p2 = 0, p3 = 0;
  #pragma unroll
  for (int h = 0; h < 16; ++h) {
    float kv = kr[h];
    p0 += kv * w0[h];
    p1 += kv * w0[512 + h];
    p2 += kv * w0[1024 + h];
    p3 += kv * w0[1536 + h];
  }
  for (int off = 16; off > 0; off >>= 1) {
    p0 += __shfl_down(p0, off);
    p1 += __shfl_down(p1, off);
    p2 += __shfl_down(p2, off);
    p3 += __shfl_down(p3, off);
  }
  if (lane == 0) {
    size_t base = (size_t)b * 4096 + t;   // [B][4][T]
    scores[base]          = (p0 + ck[0]) * SCALEF;
    scores[base + 1024]   = (p1 + ck[1]) * SCALEF;
    scores[base + 2048]   = (p2 + ck[2]) * SCALEF;
    scores[base + 3072]   = (p3 + ck[3]) * SCALEF;
  }
}

// masked softmax + renorm per (b,n); in-place on scores
__global__ void k_softmax(float* scores, const int* input_len) {
  int b = blockIdx.x >> 2, n = blockIdx.x & 3;
  float* s = scores + ((size_t)b * 4 + n) * 1024;
  int il = input_len[b];
  __shared__ float red[256];
  float vals[4], ev[4];
  float mx = -1e30f;
  #pragma unroll
  for (int q = 0; q < 4; ++q) {
    vals[q] = s[threadIdx.x + q * 256];
    mx = fmaxf(mx, vals[q]);
  }
  red[threadIdx.x] = mx; __syncthreads();
  for (int st = 128; st > 0; st >>= 1) {
    if (threadIdx.x < st) red[threadIdx.x] = fmaxf(red[threadIdx.x], red[threadIdx.x + st]);
    __syncthreads();
  }
  mx = red[0]; __syncthreads();
  float sum = 0.0f;
  #pragma unroll
  for (int q = 0; q < 4; ++q) {
    int t = threadIdx.x + q * 256;
    ev[q] = (t < il) ? __expf(vals[q] - mx) : 0.0f;
    sum += ev[q];
  }
  red[threadIdx.x] = sum; __syncthreads();
  for (int st = 128; st > 0; st >>= 1) {
    if (threadIdx.x < st) red[threadIdx.x] += red[threadIdx.x + st];
    __syncthreads();
  }
  float inv = 1.0f / red[0];
  #pragma unroll
  for (int q = 0; q < 4; ++q) s[threadIdx.x + q * 256] = ev[q] * inv;
}

// av[b][n][h] = sum_t attn[b][n][t] * values[t][b][h]
__global__ void k_av(const float* __restrict__ values, const float* __restrict__ attn,
                     float* __restrict__ av) {
  int b = blockIdx.x, h = threadIdx.x;   // 512 threads
  const float* at = attn + (size_t)b * 4096;
  float a0 = 0, a1 = 0, a2 = 0, a3 = 0;
  for (int t = 0; t < 1024; ++t) {
    float v = values[((size_t)t * 64 + b) * 512 + h];
    a0 += at[t] * v; a1 += at[1024 + t] * v;
    a2 += at[2048 + t] * v; a3 += at[3072 + t] * v;
  }
  size_t base = (size_t)b * 2048 + h;
  av[base] = a0; av[base + 512] = a1; av[base + 1024] = a2; av[base + 1536] = a3;
}

__global__ void k_head(const float* av, const float* Wv, const float* bv, float* head) {
  int b = blockIdx.x, j = threadIdx.x;
  int n = j >> 7, d = j & 127;
  const float* a = av + ((size_t)b * 4 + n) * 512;
  const float* w = Wv + (size_t)(n * 128 + d) * 512;
  float s = bv[n * 128 + d];
  for (int h = 0; h < 512; ++h) s += a[h] * w[h];
  head[(size_t)b * 512 + j] = s;
}

__global__ void k_ctx(const float* head, const float* Wmh, const float* bmh, float* ctx) {
  int b = blockIdx.x, j = threadIdx.x;
  const float* hh = head + (size_t)b * 512;
  const float* w = Wmh + (size_t)j * 512;
  float s = bmh[j];
  for (int h = 0; h < 512; ++h) s += hh[h] * w[h];
  ctx[(size_t)b * 512 + j] = s;
}

__global__ void k_ctxpart(const float* ctx, const float* Wih0, const float* bih0,
                          const float* bhh0, float* cp) {
  for (int e = blockIdx.x * blockDim.x + threadIdx.x; e < 64 * 4096;
       e += gridDim.x * blockDim.x) {
    int b = e >> 12, g = e & 4095;
    const float* c = ctx + (size_t)b * 512;
    const float* w = Wih0 + (size_t)g * 1024 + 512;
    float s = bih0[g] + bhh0[g];
    for (int h = 0; h < 512; ++h) s += c[h] * w[h];
    cp[e] = s;
  }
}

__global__ void k_ctxproj(const float* ctx, const float* Wp1, const float* bp1, float* cpj) {
  for (int e = blockIdx.x * blockDim.x + threadIdx.x; e < 64 * 512;
       e += gridDim.x * blockDim.x) {
    int b = e >> 9, j = e & 511;
    const float* c = ctx + (size_t)b * 512;
    const float* w = Wp1 + (size_t)j * 1024 + 512;
    float s = bp1[j];
    for (int h = 0; h < 512; ++h) s += c[h] * w[h];
    cpj[e] = s;
  }
}

__global__ void k_embt(const float* Wemb, const float* Wih0, float* embT) {
  for (int e = blockIdx.x * blockDim.x + threadIdx.x; e < 34 * 4096;
       e += gridDim.x * blockDim.x) {
    int v = e >> 12, g = e & 4095;
    const float* x = Wemb + (size_t)v * 512;
    const float* w = Wih0 + (size_t)g * 1024;
    float s = 0.0f;
    for (int h = 0; h < 512; ++h) s += x[h] * w[h];
    embT[e] = s;
  }
}

__global__ void k_initstate(const float* h00, const float* c00, const float* h01,
                            const float* c01, const float* h02, const float* c02,
                            bf16_t* hA0, bf16_t* hA1, bf16_t* hA2,
                            float* c0, float* c1, float* c2) {
  const int total = 65536 + 65536 + 32768;
  for (int e = blockIdx.x * blockDim.x + threadIdx.x; e < total;
       e += gridDim.x * blockDim.x) {
    if (e < 65536) {
      int b = e >> 10, j = e & 1023;
      c0[e] = c00[j];
      hA0[aidx(b, j, 32)] = (bf16_t)h00[j];
    } else if (e < 131072) {
      int e2 = e - 65536;
      int b = e2 >> 10, j = e2 & 1023;
      c1[e2] = c01[j];
      hA1[aidx(b, j, 32)] = (bf16_t)h01[j];
    } else {
      int e2 = e - 131072;
      int b = e2 >> 9, j = e2 & 511;
      c2[e2] = c02[j];
      hA2[aidx(b, j, 16)] = (bf16_t)h02[j];
    }
  }
}

// ---------------- host ----------------
extern "C" void kernel_launch(void* const* d_in, const int* in_sizes, int n_in,
                              void* d_out, int out_size, void* d_ws, size_t ws_size,
                              hipStream_t stream) {
  const float* keys     = (const float*)d_in[0];
  const float* values   = (const float*)d_in[1];
  const int*   label    = (const int*)d_in[2];
  const int*   input_len= (const int*)d_in[4];
  const float* W_emb    = (const float*)d_in[6];
  const float* b_proj2  = (const float*)d_in[7];
  const float* W_query  = (const float*)d_in[8];
  const float* b_query  = (const float*)d_in[9];
  const float* Wk       = (const float*)d_in[10];
  const float* bk       = (const float*)d_in[11];
  const float* Wq       = (const float*)d_in[12];
  const float* bq       = (const float*)d_in[13];
  const float* Wv       = (const float*)d_in[14];
  const float* bv       = (const float*)d_in[15];
  const float* W_mh     = (const float*)d_in[16];
  const float* b_mh     = (const float*)d_in[17];
  const float* W_p1     = (const float*)d_in[18];
  const float* b_p1     = (const float*)d_in[19];
  const float* Wih0     = (const float*)d_in[20];
  const float* Whh0     = (const float*)d_in[21];
  const float* bih0     = (const float*)d_in[22];
  const float* bhh0     = (const float*)d_in[23];
  const float* h00      = (const float*)d_in[24];
  const float* c00      = (const float*)d_in[25];
  const float* Wih1     = (const float*)d_in[26];
  const float* Whh1     = (const float*)d_in[27];
  const float* bih1     = (const float*)d_in[28];
  const float* bhh1     = (const float*)d_in[29];
  const float* h01      = (const float*)d_in[30];
  const float* c01      = (const float*)d_in[31];
  const float* Wih2     = (const float*)d_in[32];
  const float* Whh2     = (const float*)d_in[33];
  const float* bih2     = (const float*)d_in[34];
  const float* bhh2     = (const float*)d_in[35];
  const float* h02      = (const float*)d_in[36];
  const float* c02      = (const float*)d_in[37];

  char* ws = (char*)d_ws;
  size_t off = 0;
  auto alloc = [&](size_t bytes) -> void* {
    off = (off + 255) & ~(size_t)255;
    void* p = ws + off;
    off += bytes;
    return p;
  };

  unsigned* cnt   = (unsigned*)alloc(256);
  bf16_t* Whh0p   = (bf16_t*)alloc((size_t)4096 * 1024 * 2);
  bf16_t* Whh1p   = (bf16_t*)alloc((size_t)4096 * 1024 * 2);
  bf16_t* Wih1p   = (bf16_t*)alloc((size_t)4096 * 1024 * 2);
  bf16_t* Wih2p   = (bf16_t*)alloc((size_t)2048 * 1024 * 2);
  bf16_t* Whh2p   = (bf16_t*)alloc((size_t)2048 * 512 * 2);
  bf16_t* Wp1aP   = (bf16_t*)alloc((size_t)512 * 512 * 2);
  float*  G0      = (float*)alloc((size_t)64 * 4096 * 4);
  float*  G1      = (float*)alloc((size_t)64 * 4096 * 4);
  float*  G2      = (float*)alloc((size_t)64 * 2048 * 4);
  bf16_t* hA0     = (bf16_t*)alloc((size_t)64 * 1024 * 2);
  bf16_t* hA1     = (bf16_t*)alloc((size_t)64 * 1024 * 2);
  bf16_t* hA2     = (bf16_t*)alloc((size_t)64 * 512 * 2);
  float*  c0b     = (float*)alloc((size_t)64 * 1024 * 4);
  float*  c1b     = (float*)alloc((size_t)64 * 1024 * 4);
  float*  c2b     = (float*)alloc((size_t)64 * 512 * 4);
  float*  yF      = (float*)alloc((size_t)64 * 512 * 4);
  float*  wkq     = (float*)alloc((size_t)4 * 512 * 4);
  float*  ck      = (float*)alloc(256);
  float*  scores  = (float*)alloc((size_t)64 * 4 * 1024 * 4);
  float*  av      = (float*)alloc((size_t)64 * 4 * 512 * 4);
  float*  headB   = (float*)alloc((size_t)64 * 512 * 4);
  float*  ctxB    = (float*)alloc((size_t)64 * 512 * 4);
  float*  ctxPart = (float*)alloc((size_t)64 * 4096 * 4);
  float*  ctxProj = (float*)alloc((size_t)64 * 512 * 4);
  float*  embT    = (float*)alloc((size_t)34 * 4096 * 4);
  (void)ws_size; (void)in_sizes; (void)n_in; (void)out_size;

  k_init<<<1, 64, 0, stream>>>(cnt);

  // bf16 B-panel packing (weights stay L2-resident: ~32 MB total)
  k_pack<<<2048, 256, 0, stream>>>(Whh0, Whh0p, 4096, 1024, 1024, 0);
  k_pack<<<2048, 256, 0, stream>>>(Whh1, Whh1p, 4096, 1024, 1024, 0);
  k_pack<<<2048, 256, 0, stream>>>(Wih1, Wih1p, 4096, 1024, 1024, 0);
  k_pack<<<1024, 256, 0, stream>>>(Wih2, Wih2p, 2048, 1024, 1024, 0);
  k_pack<<<512, 256, 0, stream>>>(Whh2, Whh2p, 2048, 512, 512, 0);
  k_pack<<<128, 256, 0, stream>>>(W_p1, Wp1aP, 512, 512, 1024, 0);

  // attention chain (context is step-invariant)
  k_attn_small<<<1, 512, 0, stream>>>(h02, W_query, b_query, Wq, bq, Wk, bk, wkq, ck);
  k_scores<<<8192, 256, 0, stream>>>(keys, wkq, ck, scores);
  k_softmax<<<256, 256, 0, stream>>>(scores, input_len);
  k_av<<<64, 512, 0, stream>>>(values, scores, av);
  k_head<<<64, 512, 0, stream>>>(av, Wv, bv, headB);
  k_ctx<<<64, 512, 0, stream>>>(headB, W_mh, b_mh, ctxB);
  k_ctxpart<<<1024, 256, 0, stream>>>(ctxB, Wih0, bih0, bhh0, ctxPart);
  k_ctxproj<<<128, 256, 0, stream>>>(ctxB, W_p1, b_p1, ctxProj);
  k_embt<<<544, 256, 0, stream>>>(W_emb, Wih0, embT);
  k_initstate<<<640, 256, 0, stream>>>(h00, c00, h01, c01, h02, c02,
                                       hA0, hA1, hA2, c0b, c1b, c2b);

  PArgs a;
  a.Whh0p = Whh0p; a.Whh1p = Whh1p; a.Wih1p = Wih1p; a.Wih2p = Wih2p;
  a.Whh2p = Whh2p; a.Wp1aP = Wp1aP;
  a.G0 = G0; a.G1 = G1; a.G2 = G2;
  a.hA0 = hA0; a.hA1 = hA1; a.hA2 = hA2;
  a.c0 = c0b; a.c1 = c1b; a.c2 = c2b;
  a.yF = yF;
  a.embT = embT; a.ctxPart = ctxPart; a.ctxProj = ctxProj;
  a.bih1 = bih1; a.bhh1 = bhh1; a.bih2 = bih2; a.bhh2 = bhh2;
  a.Wemb = W_emb; a.bproj2 = b_proj2;
  a.label = label;
  a.out = (float*)d_out;
  a.cnt = cnt;

  las_persistent<<<NWG, TPB, 0, stream>>>(a);
}
